// PhysicsGuidedAttentionCorrected_6098853560432
// MI455X (gfx1250) — matrix operand order repaired and checked by reading, not compiled, not run on hardware
//
#include <hip/hip_runtime.h>
#include <cstdint>

typedef __attribute__((ext_vector_type(16))) _Float16 v16h;
typedef __attribute__((ext_vector_type(8)))  float    v8f;
typedef int i32x2 __attribute__((vector_size(8)));
typedef int i32x4 __attribute__((vector_size(16)));

#define DEV __device__ __forceinline__

constexpr int Bn = 8, Nn = 1024, Cn = 256, NH = 8, HD = 32;
constexpr float SCALE = 0.17677669529663687f;   // 32^-0.5 folded into Q
constexpr float L2E   = 1.4426950408889634f;    // log2(e): softmax in base-2

// ---------- async global->LDS copies (ASYNCcnt path), with safe fallback ----
#if defined(__gfx1250__) && \
    __has_builtin(__builtin_amdgcn_global_load_async_to_lds_b64) && \
    __has_builtin(__builtin_amdgcn_global_load_async_to_lds_b128) && \
    __has_builtin(__builtin_amdgcn_s_wait_asynccnt)
#define HAVE_ASYNC_LDS 1
#else
#define HAVE_ASYNC_LDS 0
#endif

DEV void async_copy_b64(void* lds, const void* g) {
#if HAVE_ASYNC_LDS
  __builtin_amdgcn_global_load_async_to_lds_b64(
      (__attribute__((address_space(1))) i32x2*)(uintptr_t)g,
      (__attribute__((address_space(3))) i32x2*)(uintptr_t)lds, 0, 0);
#else
  *reinterpret_cast<uint2*>(lds) = *reinterpret_cast<const uint2*>(g);
#endif
}

DEV void async_copy_b128(void* lds, const void* g) {
#if HAVE_ASYNC_LDS
  __builtin_amdgcn_global_load_async_to_lds_b128(
      (__attribute__((address_space(1))) i32x4*)(uintptr_t)g,
      (__attribute__((address_space(3))) i32x4*)(uintptr_t)lds, 0, 0);
#else
  *reinterpret_cast<uint4*>(lds) = *reinterpret_cast<const uint4*>(g);
#endif
}

DEV void async_wait0() {
#if HAVE_ASYNC_LDS
  __builtin_amdgcn_s_wait_asynccnt(0);
#endif
}

// ---------- WMMA helpers (layouts per CDNA5 ISA §7.12.2) ----------

DEV v8f wmma(v16h a, v16h b, v8f c) {
  return __builtin_amdgcn_wmma_f32_16x16x32_f16(false, a, false, b, (short)0, c,
                                                false, false);
}

// 16-bit A-matrix 16x32: lane row = lane&15, grp = lane>>4,
// this code's k order (same for A and B): element e -> K = grp*16 + e  => one contiguous 16-half piece.
DEV v16h load_afrag(const _Float16* rowp_k0, int grp) {
  union { uint4 u[2]; v16h h; } t;
  t.u[0] = *reinterpret_cast<const uint4*>(rowp_k0 + grp * 16);
  t.u[1] = *reinterpret_cast<const uint4*>(rowp_k0 + grp * 16 + 8);
  return t.h;
}

// 16-bit B-matrix 32x16: lane col = lane&15, grp = lane>>4,
// this code's k order (same as A): element e -> K = grp*16 + e  => one contiguous 16-half piece.
DEV v16h load_bfrag(const _Float16* p) {
  union { uint4 u[2]; v16h h; } t;
  t.u[0] = *reinterpret_cast<const uint4*>(p);
  t.u[1] = *reinterpret_cast<const uint4*>(p + 8);
  return t.h;
}

// ---------- prep kernels ----------

__global__ void cvt_f16_kernel(const float* __restrict__ src,
                               _Float16* __restrict__ dst, int n) {
  int i = blockIdx.x * blockDim.x + threadIdx.x;
  if (i < n) dst[i] = (_Float16)src[i];
}

__global__ void wind_kernel(const float* __restrict__ u,
                            const float* __restrict__ v,
                            float* __restrict__ wstr) {
  int i = blockIdx.x * blockDim.x + threadIdx.x;   // over B*N
  if (i >= Bn * Nn) return;
  int b = i >> 10, n = i & 1023;
  int r = n >> 5, c = n & 31;
  const float* ub = u + (size_t)b * 64 * 64;
  const float* vb = v + (size_t)b * 64 * 64;
  float acc = 0.f;
#pragma unroll
  for (int dy = 0; dy < 2; ++dy)
#pragma unroll
    for (int dx = 0; dx < 2; ++dx) {
      int idx = (2 * r + dy) * 64 + (2 * c + dx);
      float uu = ub[idx], vv = vb[idx];
      acc += sqrtf(uu * uu + vv * vv + 1e-8f);
    }
  wstr[i] = 0.25f * acc;
}

// ---------- QKV projection: [8192,256] x [256,768]^T -> q,k,v (f16) ----------
// grid (512, 3), block 256. Workgroup: 16 rows x 256 cols; wave: 16x32.

__global__ __launch_bounds__(256) void qkv_gemm_kernel(
    const _Float16* __restrict__ xh, const _Float16* __restrict__ wh,
    _Float16* __restrict__ qh, _Float16* __restrict__ kh,
    _Float16* __restrict__ vt) {
  __shared__ _Float16 sA[16 * Cn];                   // 8 KB x-tile
  const int tid = threadIdx.x, lane = tid & 31, w = tid >> 5;
  const int col = lane & 15, grp = lane >> 4;
  const int m0 = blockIdx.x * 16;

  {  // async DMA stage of the shared 16x256 A tile (32B per thread)
    int row = tid >> 4, cb = (tid & 15) * 16;
    const _Float16* g = xh + (size_t)(m0 + row) * Cn + cb;
    async_copy_b128(&sA[row * Cn + cb], g);
    async_copy_b128(&sA[row * Cn + cb + 8], g + 8);
  }
  async_wait0();
  __syncthreads();

  const int j0 = blockIdx.y * 256 + w * 32;
  v8f c0 = {}, c1 = {};
  for (int k0 = 0; k0 < Cn; k0 += 32) {
    v16h a  = load_afrag(&sA[col * Cn + k0], grp);
    v16h b0 = load_bfrag(&wh[(size_t)(j0 + col) * Cn + k0 + grp * 16]);
    v16h b1 = load_bfrag(&wh[(size_t)(j0 + 16 + col) * Cn + k0 + grp * 16]);
    c0 = wmma(a, b0, c0);
    c1 = wmma(a, b1, c1);
  }

#pragma unroll
  for (int r = 0; r < 8; ++r) {
    int m = m0 + r + 8 * grp;
    int b = m >> 10, n = m & 1023;
#pragma unroll
    for (int t = 0; t < 2; ++t) {
      int j = j0 + t * 16 + col;
      float val = (t == 0) ? c0[r] : c1[r];
      int s = j >> 8, rem = j & 255, h = rem >> 5, d = rem & 31;
      if (s == 0)                                      // Q (pre-scaled)
        qh[(((size_t)b * NH + h) * Nn + n) * HD + d] = (_Float16)(val * SCALE);
      else if (s == 1)                                 // K row-major [n][d]
        kh[(((size_t)b * NH + h) * Nn + n) * HD + d] = (_Float16)val;
      else                                             // V transposed [d][n]
        vt[(((size_t)b * NH + h) * HD + d) * Nn + n] = (_Float16)val;
    }
  }
}

// ---------- bias (transcendental-free inner form) ----------
// sigmoid((wi+wj)/2 - 5) == 1 / (1 + e^{-wi/2} * e^{5-wj/2})

DEV float bias_fast(float ei, float ej, float arw, float bcw,
                    float alpha, float beta) {
  float rel = (ej - ei) * 1e-3f;
  rel = rel > 0.f ? rel : 0.f;
  float wf = 1.0f / (1.0f + arw * bcw);
  float bias = -alpha * rel * (1.0f - beta * wf);
  bias = bias < -10.f ? -10.f : bias;
  return bias > 0.f ? 0.f : bias;
}

// ---------- fused flash attention with elevation/wind bias ----------
// grid (8, 64): x = query block of 128 rows, y = (b,h). block 256 = 8 waves.
// Double-buffered async K/V staging: wait own ASYNCcnt, barrier, issue next.

__global__ __launch_bounds__(256) void attn_kernel(
    const _Float16* __restrict__ qh, const _Float16* __restrict__ kh,
    const _Float16* __restrict__ vt, const float* __restrict__ elev,
    const float* __restrict__ wstr, const float* __restrict__ alpha_p,
    const float* __restrict__ beta_p, _Float16* __restrict__ attnh) {
  __shared__ _Float16 sK[2][32 * 32];     // [buf][key][hd]   4 KB
  __shared__ _Float16 sV[2][32 * 32];     // [buf][d][key]    4 KB
  __shared__ _Float16 sP[8][16 * 32];     // per-wave P scratch, 8 KB

  const int tid = threadIdx.x, lane = tid & 31, w = tid >> 5;
  const int col = lane & 15, grp = lane >> 4;
  const int bh = blockIdx.y, b = bh >> 3, h = bh & 7;
  const float alpha = *alpha_p, beta = *beta_p;

  const _Float16* Q = qh + (size_t)bh * Nn * HD;
  const _Float16* K = kh + (size_t)bh * Nn * HD;
  const _Float16* V = vt + (size_t)bh * HD * Nn;
  const float* eb = elev + (size_t)b * Nn;
  const float* wb = wstr + (size_t)b * Nn;

  const int qbase = blockIdx.x * 128 + w * 16;

  v16h qa = load_afrag(Q + (size_t)(qbase + col) * HD, grp);

  float ei[8], ar[8], m_i[8], l_i[8];
  v8f o0 = {}, o1 = {};
#pragma unroll
  for (int r = 0; r < 8; ++r) {
    int row = qbase + r + 8 * grp;        // matches C-matrix row layout
    ei[r] = eb[row];
    ar[r] = expf(-0.5f * wb[row]);        // e^{-wi/2}, once per row
    m_i[r] = -1e30f; l_i[r] = 0.f;
  }

  // staging slice for this thread: 8 bytes of K and 8 of V per chunk
  const int sr = tid >> 3, scp = (tid & 7) * 4;
  auto issue_chunk = [&](int j0, int buf) {
    async_copy_b64(&sK[buf][sr * 32 + scp], &K[(size_t)(j0 + sr) * HD + scp]);
    async_copy_b64(&sV[buf][sr * 32 + scp], &V[(size_t)sr * Nn + j0 + scp]);
  };

  issue_chunk(0, 0);

  for (int it = 0; it < Nn / 32; ++it) {
    const int j0 = it * 32, buf = it & 1;
    async_wait0();                        // my async writes for chunk `it` done
    __syncthreads();                      // => everyone's writes visible
    if (it + 1 < Nn / 32) issue_chunk(j0 + 32, buf ^ 1);  // overlap next DMA

    // S = Q K^T (scale pre-folded into Q); two 16-key tiles
    v8f s0 = {}, s1 = {};
    {
      v16h kb0 = load_bfrag(&sK[buf][col * 32 + grp * 16]);
      v16h kb1 = load_bfrag(&sK[buf][(16 + col) * 32 + grp * 16]);
      s0 = wmma(qa, kb0, s0);
      s1 = wmma(qa, kb1, s1);
    }

    int jc0 = j0 + col, jc1 = jc0 + 16;
    float ej0 = eb[jc0], ej1 = eb[jc1];
    float bc0 = expf(5.0f - 0.5f * wb[jc0]);   // e^{5-wj/2}, once per chunk
    float bc1 = expf(5.0f - 0.5f * wb[jc1]);

#pragma unroll
    for (int r = 0; r < 8; ++r) {
      // scores to base-2 domain; bias is pure FMA/rcp
      float t0 = (s0[r] + bias_fast(ei[r], ej0, ar[r], bc0, alpha, beta)) * L2E;
      float t1 = (s1[r] + bias_fast(ei[r], ej1, ar[r], bc1, alpha, beta)) * L2E;
      float rm = fmaxf(t0, t1);
      rm = fmaxf(rm, __shfl_xor(rm, 1));
      rm = fmaxf(rm, __shfl_xor(rm, 2));
      rm = fmaxf(rm, __shfl_xor(rm, 4));
      rm = fmaxf(rm, __shfl_xor(rm, 8));
      float mnew = fmaxf(m_i[r], rm);
      float scl = exp2f(m_i[r] - mnew);        // native v_exp_f32
      m_i[r] = mnew;
      float e0 = exp2f(t0 - mnew), e1 = exp2f(t1 - mnew);
      float rs = e0 + e1;
      rs += __shfl_xor(rs, 1);
      rs += __shfl_xor(rs, 2);
      rs += __shfl_xor(rs, 4);
      rs += __shfl_xor(rs, 8);
      l_i[r] = l_i[r] * scl + rs;
      o0[r] *= scl; o1[r] *= scl;
      // C-layout -> LDS row-major for the A-fragment reload below
      sP[w][(r + 8 * grp) * 32 + col]      = (_Float16)e0;
      sP[w][(r + 8 * grp) * 32 + 16 + col] = (_Float16)e1;
    }

    // O += P (16x32) * V (32x16 per hd half); same-wave DS ordering holds
    {
      v16h pa  = load_afrag(&sP[w][col * 32], grp);
      v16h vb0 = load_bfrag(&sV[buf][col * 32 + grp * 16]);        // d 0..15
      v16h vb1 = load_bfrag(&sV[buf][(16 + col) * 32 + grp * 16]); // d 16..31
      o0 = wmma(pa, vb0, o0);
      o1 = wmma(pa, vb1, o1);
    }
  }

#pragma unroll
  for (int r = 0; r < 8; ++r) {
    int row = qbase + r + 8 * grp;
    float inv = 1.0f / l_i[r];
    size_t base = ((size_t)b * Nn + row) * Cn + h * HD;
    attnh[base + col]      = (_Float16)(o0[r] * inv);
    attnh[base + 16 + col] = (_Float16)(o1[r] * inv);
  }
}

// ---------- output projection: [8192,256] x [256,256]^T + bias -> f32 ----------

__global__ __launch_bounds__(256) void proj_gemm_kernel(
    const _Float16* __restrict__ ah, const _Float16* __restrict__ wh,
    const float* __restrict__ bproj, float* __restrict__ out) {
  __shared__ _Float16 sA[16 * Cn];
  const int tid = threadIdx.x, lane = tid & 31, w = tid >> 5;
  const int col = lane & 15, grp = lane >> 4;
  const int m0 = blockIdx.x * 16;

  {
    int row = tid >> 4, cb = (tid & 15) * 16;
    const _Float16* g = ah + (size_t)(m0 + row) * Cn + cb;
    async_copy_b128(&sA[row * Cn + cb], g);
    async_copy_b128(&sA[row * Cn + cb + 8], g + 8);
  }
  async_wait0();
  __syncthreads();

  const int j0 = w * 32;
  v8f c0 = {}, c1 = {};
  for (int k0 = 0; k0 < Cn; k0 += 32) {
    v16h a  = load_afrag(&sA[col * Cn + k0], grp);
    v16h b0 = load_bfrag(&wh[(size_t)(j0 + col) * Cn + k0 + grp * 16]);
    v16h b1 = load_bfrag(&wh[(size_t)(j0 + 16 + col) * Cn + k0 + grp * 16]);
    c0 = wmma(a, b0, c0);
    c1 = wmma(a, b1, c1);
  }

  float bb0 = bproj[j0 + col], bb1 = bproj[j0 + 16 + col];
#pragma unroll
  for (int r = 0; r < 8; ++r) {
    int m = m0 + r + 8 * grp;
    out[(size_t)m * Cn + j0 + col]      = c0[r] + bb0;
    out[(size_t)m * Cn + j0 + 16 + col] = c1[r] + bb1;
  }
}

// ---------- launcher ----------

extern "C" void kernel_launch(void* const* d_in, const int* in_sizes, int n_in,
                              void* d_out, int out_size, void* d_ws, size_t ws_size,
                              hipStream_t stream) {
  const float* x     = (const float*)d_in[0];
  const float* elev  = (const float*)d_in[1];
  const float* uw    = (const float*)d_in[2];
  const float* vw    = (const float*)d_in[3];
  const float* wqkv  = (const float*)d_in[4];
  const float* wproj = (const float*)d_in[5];
  const float* bproj = (const float*)d_in[6];
  const float* alpha = (const float*)d_in[7];
  const float* beta  = (const float*)d_in[8];
  float* out = (float*)d_out;

  // workspace carve-up (~21.5 MB total)
  char* p = (char*)d_ws;
  _Float16* xh     = (_Float16*)p; p += (size_t)Bn * Nn * Cn * 2;       // 4 MB
  _Float16* wqkvh  = (_Float16*)p; p += (size_t)3 * Cn * Cn * 2;        // 384 KB
  _Float16* wprojh = (_Float16*)p; p += (size_t)Cn * Cn * 2;            // 128 KB
  float*    wstr   = (float*)p;    p += (size_t)Bn * Nn * 4;            // 32 KB
  _Float16* qh     = (_Float16*)p; p += (size_t)Bn * NH * Nn * HD * 2;  // 4 MB
  _Float16* kh     = (_Float16*)p; p += (size_t)Bn * NH * Nn * HD * 2;  // 4 MB
  _Float16* vt     = (_Float16*)p; p += (size_t)Bn * NH * HD * Nn * 2;  // 4 MB
  _Float16* attnh  = (_Float16*)p;                                      // 4 MB

  const int nx  = Bn * Nn * Cn;         // 2,097,152
  const int nwq = 3 * Cn * Cn;          // 196,608
  const int nwp = Cn * Cn;              // 65,536

  cvt_f16_kernel<<<(nx + 255) / 256, 256, 0, stream>>>(x, xh, nx);
  cvt_f16_kernel<<<(nwq + 255) / 256, 256, 0, stream>>>(wqkv, wqkvh, nwq);
  cvt_f16_kernel<<<(nwp + 255) / 256, 256, 0, stream>>>(wproj, wprojh, nwp);
  wind_kernel<<<(Bn * Nn + 255) / 256, 256, 0, stream>>>(uw, vw, wstr);

  qkv_gemm_kernel<<<dim3(512, 3), 256, 0, stream>>>(xh, wqkvh, qh, kh, vt);
  attn_kernel<<<dim3(Nn / 128, Bn * NH), 256, 0, stream>>>(
      qh, kh, vt, elev, wstr, alpha, beta, attnh);
  proj_gemm_kernel<<<512, 256, 0, stream>>>(attnh, wprojh, bproj, out);
}
